// MalConvLowMem_75144747811159
// MI455X (gfx1250) — matrix operand order repaired and checked by reading, not compiled, not run on hardware
//
#include <hip/hip_runtime.h>

// ---------------------------------------------------------------------------
// MalConv low-mem gated conv + max-over-time for MI455X (gfx1250, wave32).
//
// Closed-form position set (replayed from the reference chunk loop):
//   chunks i = 0..31, chunk start = 63489*i
//   i < 31  -> 125 windows at start + 512*k
//   i == 31 ->  62 windows
//   NP = 31*125 + 62 = 3937 unique, unclipped positions.
//
// GEMM view: A[m, r] = x[b, pos_m + r/8, r%8]  (r = k*8+e, contiguous in x)
//            B[r, n] = wT[c_n][r]              (pre-transposed f16 weights)
// D = A*B accumulated f32 via v_wmma_f32_16x16x32_f16, then GLU + max.
// ---------------------------------------------------------------------------

typedef __attribute__((ext_vector_type(16))) _Float16 v16h;
typedef __attribute__((ext_vector_type(8)))  float    v8f;
typedef __attribute__((ext_vector_type(4)))  float    v4f;

#define T_LEN    2000000L
#define NPOS_M1  3936      // NP - 1
#define NFULL    3875      // 31 * 125
#define WELEMS   524288    // C * E * K = 128 * 4096

// ---------------------------------------------------------------------------
// Prep: wT[m][c][r] = (f16) w_m[c, r%8, r/8]  (r = k*8 + e), and out = -inf.
// ---------------------------------------------------------------------------
__global__ void malconv_prep_kernel(const float* __restrict__ w1,
                                    const float* __restrict__ w2,
                                    _Float16* __restrict__ wT,
                                    float* __restrict__ out) {
    unsigned gid = blockIdx.x * blockDim.x + threadIdx.x;   // 0 .. 2*WELEMS-1
    unsigned m   = gid >> 19;            // which weight tensor
    unsigned rem = gid & (WELEMS - 1u);  // c*4096 + r
    unsigned c   = rem >> 12;
    unsigned r   = rem & 4095u;
    const float* w = m ? w2 : w1;
    // w[c, e, k] at c*4096 + e*512 + k ; here e = r&7, k = r>>3
    wT[gid] = (_Float16)w[c * 4096u + (r & 7u) * 512u + (r >> 3)];
    if (gid < 512u) out[gid] = -__builtin_inff();
}

// ---------------------------------------------------------------------------
// CAS-based float atomic max (safe for any sign; out pre-initialized to -inf).
// ---------------------------------------------------------------------------
__device__ __forceinline__ void atomic_fmax(float* addr, float val) {
    unsigned int* ua  = (unsigned int*)addr;
    unsigned int  cur = __float_as_uint(*addr);
    while (__uint_as_float(cur) < val) {
        unsigned int prev = atomicCAS(ua, cur, __float_as_uint(val));
        if (prev == cur) break;
        cur = prev;
    }
}

// ---------------------------------------------------------------------------
// Main: grid = (62, B), block = 128 (4 waves). Each wave: one 16-position
// tile x 128 channels x both gates (16 f32 WMMA accumulators).
// ---------------------------------------------------------------------------
__global__ void __launch_bounds__(128) malconv_glu_max_kernel(
    const float* __restrict__ x,
    const _Float16* __restrict__ wT,
    const float* __restrict__ b1,
    const float* __restrict__ b2,
    float* __restrict__ out)
{
    const int lane   = threadIdx.x & 31;
    const int wave   = threadIdx.x >> 5;
    const int b      = blockIdx.y;
    const int row    = lane & 15;   // matrix row (A/M) / column (B,N=c offset)
    const int rowsel = lane >> 4;   // low/high half of the wave

    // Flat window index for this lane's A-row; clamp ragged tail (dups are
    // harmless under max and keep EXEC all-ones for the WMMAs).
    int j = (blockIdx.x * 4 + wave) * 16 + row;
    j = j > NPOS_M1 ? NPOS_M1 : j;
    const int  big = (j >= NFULL);
    const int  ci  = big ? 31 : (j / 125);
    const int  ck  = big ? (j - NFULL) : (j - ci * 125);
    const long pos = 63489L * (long)ci + 512L * (long)ck;

    const float*    xrow = x + ((long)b * T_LEN + pos) * 8L;  // 4096 contiguous f32
    const _Float16* wt1  = wT;
    const _Float16* wt2  = wT + WELEMS;

    v8f acc1[8] = {};   // c1 for channel tiles 0..7
    v8f acc2[8] = {};   // c2 for channel tiles 0..7

    for (int s = 0; s < 128; ++s) {        // 4096 / 32 k-steps
        // ---- A fragment (16x32 f16). Lane-low holds K = 0..15, lane-high
        // K = 16..31 (matches B) => rows k = 4s+2*rowsel and k = 4s+2*rowsel+1,
        // e = 0..7 fastest (contiguous in x). Nontemporal: x has zero reuse.
        const float* pa = xrow + (size_t)(4 * s + 2 * rowsel) * 8;
        v4f a0 = __builtin_nontemporal_load((const v4f*)pa);
        v4f a1 = __builtin_nontemporal_load((const v4f*)pa + 1);
        v4f a2 = __builtin_nontemporal_load((const v4f*)(pa + 8));
        v4f a3 = __builtin_nontemporal_load((const v4f*)(pa + 8) + 1);
        v16h A;
#pragma unroll
        for (int i = 0; i < 4; ++i) {
            A[i]      = (_Float16)a0[i];
            A[4 + i]  = (_Float16)a1[i];
            A[8 + i]  = (_Float16)a2[i];
            A[12 + i] = (_Float16)a3[i];
        }

        // ---- B fragments (32x16 f16). Lane-low holds K=0..15 contiguously,
        // lane-high K=16..31 => one 32B contiguous load from wT per tile.
        const int rbase = 32 * s + rowsel * 16;
#pragma unroll
        for (int t = 0; t < 8; ++t) {
            const size_t woff = (size_t)(t * 16 + row) * 4096 + rbase;
            v16h B1 = *(const v16h*)(wt1 + woff);
            acc1[t] = __builtin_amdgcn_wmma_f32_16x16x32_f16(
                false, A, false, B1, (short)0, acc1[t], false, false);
            v16h B2 = *(const v16h*)(wt2 + woff);
            acc2[t] = __builtin_amdgcn_wmma_f32_16x16x32_f16(
                false, A, false, B2, (short)0, acc2[t], false, false);
        }
    }

    // ---- Epilogue: bias + GLU + max. C/D layout: VGPR v, lanes 0-15 ->
    // (M=v, N=lane); lanes 16-31 -> (M=v+8, N=lane-16). So each lane owns one
    // channel (c = 16t + row) across 8 position-rows; fold halves via shfl.
#pragma unroll
    for (int t = 0; t < 8; ++t) {
        const int   c     = t * 16 + row;
        const float bias1 = b1[c];
        const float bias2 = b2[c];
        float m = -__builtin_inff();
#pragma unroll
        for (int v = 0; v < 8; ++v) {
            float z1 = acc1[t][v] + bias1;
            float z2 = acc2[t][v] + bias2;
            float g  = z1 * (1.0f / (1.0f + __expf(-z2)));
            m = fmaxf(m, g);
        }
        m = fmaxf(m, __shfl_xor(m, 16, 32));   // combine rows m and m+8
        if (rowsel == 0) atomic_fmax(out + b * 128 + c, m);
    }
}

// ---------------------------------------------------------------------------
extern "C" void kernel_launch(void* const* d_in, const int* in_sizes, int n_in,
                              void* d_out, int out_size, void* d_ws, size_t ws_size,
                              hipStream_t stream) {
    (void)in_sizes; (void)n_in; (void)out_size; (void)ws_size;
    const float* x  = (const float*)d_in[0];
    const float* w1 = (const float*)d_in[1];
    const float* b1 = (const float*)d_in[2];
    const float* w2 = (const float*)d_in[3];
    const float* b2 = (const float*)d_in[4];
    float*    out = (float*)d_out;
    _Float16* wT  = (_Float16*)d_ws;   // 2 * 524288 f16 = 2 MB scratch

    // 1) weight transpose->f16 + out = -inf (re-run every call: graph-safe).
    malconv_prep_kernel<<<2 * WELEMS / 256, 256, 0, stream>>>(w1, w2, wT, out);

    // 2) WMMA GEMM + GLU + max. 62*4 waves/b cover ceil(3937/16)=247 p-tiles.
    dim3 grid(62, 4, 1);
    malconv_glu_max_kernel<<<grid, 128, 0, stream>>>(x, wT, b1, b2, out);
}
